// CLIPCrossAttention_54468775248334
// MI455X (gfx1250) — matrix operand order repaired and checked by reading, not compiled, not run on hardware
//
#include <hip/hip_runtime.h>
#include <hip/hip_bf16.h>

#ifndef __has_builtin
#define __has_builtin(x) 0
#endif

#if __has_builtin(__builtin_amdgcn_global_load_async_to_lds_b128)
#define HAVE_ASYNC_LDS 1
#else
#define HAVE_ASYNC_LDS 0
#endif

typedef int v4i_ __attribute__((ext_vector_type(4)));
typedef __attribute__((address_space(1))) v4i_* gv4p;   // global int4*
typedef __attribute__((address_space(3))) v4i_* lv4p;   // LDS int4*

#if HAVE_ASYNC_LDS
__device__ __forceinline__ void async_cp16(const void* g, void* l) {
    __builtin_amdgcn_global_load_async_to_lds_b128((gv4p)g, (lv4p)l, 0, 0);
}
#if __has_builtin(__builtin_amdgcn_s_wait_asynccnt)
#define WAIT_ASYNC(n) __builtin_amdgcn_s_wait_asynccnt(n)
#else
#define WAIT_ASYNC(n) asm volatile("s_wait_asynccnt %0" ::"i"(n))
#endif
#else
__device__ __forceinline__ void async_cp16(const void* g, void* l) {
    *(uint4*)l = *(const uint4*)g;
}
#define WAIT_ASYNC(n)
#endif

typedef __attribute__((ext_vector_type(16))) __bf16 v16bf;
typedef __attribute__((ext_vector_type(8)))  float  v8f;

union Frag { unsigned u[8]; v16bf v; };

__device__ __forceinline__ unsigned short f2bf(float f) {
    union { float f; unsigned u; } c; c.f = f;
    unsigned u = c.u;
    u += 0x7FFFu + ((u >> 16) & 1u);          // round-to-nearest-even
    return (unsigned short)(u >> 16);
}
// 16-bit A-matrix 16x32 fragment: pair base k for element i, lane-half hf
__device__ __forceinline__ int a_koff(int i, int hf) {
    return (hf ? 8 : 0) + (i < 4 ? 2 * i : 16 + 2 * (i - 4));
}
// 16-bit B-matrix 32x16 fragment: pair base k for element i, lane-half hf
__device__ __forceinline__ int b_koff(int i, int hf) {
    return (hf ? 8 : 0) + (i < 4 ? 2 * i : 16 + 2 * (i - 4));
}
__device__ __forceinline__ v8f wmma_bf16(const Frag& a, const Frag& b, v8f c) {
    return __builtin_amdgcn_wmma_f32_16x16x32_bf16(
        false, a.v, false, b.v, (short)0, c, false, false);
}
__device__ __forceinline__ v8f vzero8() {
    v8f z = {0.f, 0.f, 0.f, 0.f, 0.f, 0.f, 0.f, 0.f};
    return z;
}

static constexpr int Hd  = 1024;   // hidden
static constexpr int NHh = 16;     // heads
static constexpr int HDd = 64;     // head dim
static constexpr int Bb  = 8;
static constexpr int Ss  = 1024;
static constexpr int Mm  = Bb * Ss;        // 8192 rows
static constexpr int ADP = 64;

// ---------------- LayerNorm + fp32 -> bf16 ----------------
__global__ __launch_bounds__(256)
void ln_bf16_kernel(const float* __restrict__ x, const float* __restrict__ gamma,
                    const float* __restrict__ beta, unsigned short* __restrict__ y) {
    __shared__ float r1[256];
    __shared__ float r2[256];
    const int row = blockIdx.x;
    const int tid = threadIdx.x;
    float v[4];
    float s = 0.f, s2 = 0.f;
    #pragma unroll
    for (int t = 0; t < 4; ++t) {
        int c = t * 256 + tid;
        float xv = x[(size_t)row * Hd + c];
        v[t] = xv; s += xv; s2 += xv * xv;
    }
    r1[tid] = s; r2[tid] = s2;
    __syncthreads();
    for (int off = 128; off > 0; off >>= 1) {
        if (tid < off) { r1[tid] += r1[tid + off]; r2[tid] += r2[tid + off]; }
        __syncthreads();
    }
    const float mu  = r1[0] * (1.0f / Hd);
    const float var = r2[0] * (1.0f / Hd) - mu * mu;
    const float rs  = rsqrtf(var + 1e-5f);
    #pragma unroll
    for (int t = 0; t < 4; ++t) {
        int c = t * 256 + tid;
        y[(size_t)row * Hd + c] = f2bf((v[t] - mu) * rs * gamma[c] + beta[c]);
    }
}

// ---------------- Tiled transpose + fp32->bf16: Wt[C][R] = bf16(W[R][C]^T) ----------------
__global__ __launch_bounds__(256)
void transpose_conv_kernel(const float* __restrict__ W, unsigned short* __restrict__ Wt,
                           int R, int C) {
    __shared__ unsigned short t_s[64 * 65];
    const int tid = threadIdx.x;
    const int c0 = blockIdx.x * 64;
    const int r0 = blockIdx.y * 64;
    for (int e = tid; e < 4096; e += 256) {
        int r = e >> 6, c = e & 63;
        t_s[r * 65 + c] = f2bf(W[(size_t)(r0 + r) * C + c0 + c]);
    }
    __syncthreads();
    for (int e = tid; e < 4096; e += 256) {
        int c = e >> 6, r = e & 63;
        Wt[(size_t)(c0 + c) * R + r0 + r] = t_s[r * 65 + c];
    }
}

// ---------------- GEMM: C(bf16) = A(bf16, MxK) @ Wt^T + bias, Wt is N x K bf16 ----------------
// WG tile 128x128, 8 waves of 32x64, BK=32, double-buffered async LDS staging
__global__ __launch_bounds__(256)
void gemm_bias_bf16_kernel(const unsigned short* __restrict__ A,
                           const unsigned short* __restrict__ Wt,
                           const float* __restrict__ bias,
                           unsigned short* __restrict__ C) {
    constexpr int BM = 128, BN = 128, BK = 32, K = Hd, N = Hd;
    __shared__ __align__(16) unsigned short a_s[2][BM * BK];  // [row][k]
    __shared__ __align__(16) unsigned short b_s[2][BN * BK];  // [n][k]
    const int tid  = threadIdx.x;
    const int lane = tid & 31;
    const int w    = tid >> 5;
    const int hf   = (lane >= 16) ? 1 : 0;
    const int ln15 = lane & 15;
    const int wm = (w & 3) * 32;
    const int wn = (w >> 2) * 64;
    const int m_base = blockIdx.x * BM;
    const int n_base = blockIdx.y * BN;

    const unsigned short* gA = A  + (size_t)m_base * K;   // 128 rows, stride K
    const unsigned short* gB = Wt + (size_t)n_base * K;   // 128 rows, stride K

    v8f acc[2][4];
    #pragma unroll
    for (int mi = 0; mi < 2; ++mi)
        #pragma unroll
        for (int nj = 0; nj < 4; ++nj) acc[mi][nj] = vzero8();

    // stage one 128x32 tile pair (A+B): 512 chunks of 16B each, 2 per thread per matrix
    auto stage = [&](int buf, int k0) {
        #pragma unroll
        for (int p = 0; p < 2; ++p) {
            int cch = tid + p * 256;
            int row = cch >> 2, cp = cch & 3;
            async_cp16(gA + (size_t)row * K + k0 + cp * 8, &a_s[buf][row * 32 + cp * 8]);
        }
        #pragma unroll
        for (int p = 0; p < 2; ++p) {
            int cch = tid + p * 256;
            int row = cch >> 2, cp = cch & 3;
            async_cp16(gB + (size_t)row * K + k0 + cp * 8, &b_s[buf][row * 32 + cp * 8]);
        }
    };

    constexpr int NT = K / BK;          // 32 K-tiles
    stage(0, 0);
    #pragma unroll 1
    for (int it = 0; it < NT; ++it) {
        const int nk = (it + 1 < NT) ? (it + 1) * BK : it * BK;  // redundant last prefetch
        stage((it + 1) & 1, nk);
        WAIT_ASYNC(4);                  // current tile's 4 async ops complete
        __syncthreads();

        const unsigned* la = (const unsigned*)a_s[it & 1];
        const unsigned* lb = (const unsigned*)b_s[it & 1];
        Frag afr[2], bfr[4];
        #pragma unroll
        for (int mi = 0; mi < 2; ++mi) {
            int row = wm + mi * 16 + ln15;
            #pragma unroll
            for (int i = 0; i < 8; ++i)
                afr[mi].u[i] = la[row * 16 + (a_koff(i, hf) >> 1)];
        }
        #pragma unroll
        for (int nj = 0; nj < 4; ++nj) {
            int n = wn + nj * 16 + ln15;
            #pragma unroll
            for (int i = 0; i < 8; ++i)
                bfr[nj].u[i] = lb[n * 16 + (b_koff(i, hf) >> 1)];
        }
        #pragma unroll
        for (int mi = 0; mi < 2; ++mi)
            #pragma unroll
            for (int nj = 0; nj < 4; ++nj)
                acc[mi][nj] = wmma_bf16(afr[mi], bfr[nj], acc[mi][nj]);
        __syncthreads();                // done reading this buffer
    }
    // epilogue: + bias, -> bf16
    #pragma unroll
    for (int mi = 0; mi < 2; ++mi)
        #pragma unroll
        for (int nj = 0; nj < 4; ++nj)
            #pragma unroll
            for (int i = 0; i < 8; ++i) {
                int m = m_base + wm + mi * 16 + i + 8 * hf;
                int n = n_base + wn + nj * 16 + ln15;
                C[(size_t)m * N + n] = f2bf(acc[mi][nj][i] + bias[n]);
            }
}

// ---------------- Flash attention over one head, 64 query rows / WG ----------------
__global__ __launch_bounds__(128)
void attention_kernel(const unsigned short* __restrict__ Q,
                      const unsigned short* __restrict__ Kb,
                      const unsigned short* __restrict__ Vb,
                      float* __restrict__ out) {
    __shared__ __align__(16) unsigned short k_s[64 * 64];   // [kv][d]
    __shared__ __align__(16) unsigned short v_s[64 * 64];   // [d][kv] (transposed)
    __shared__ float    s_s[4][16 * 64];
    __shared__ unsigned p_s[4][16 * 32];
    __shared__ float    rscale[4][16];
    __shared__ float    rinvl[4][16];

    const int bh = blockIdx.x;                 // b*NH + h
    const int b  = bh >> 4;
    const int h  = bh & 15;
    const int tid  = threadIdx.x;
    const int lane = tid & 31;
    const int w    = tid >> 5;
    const int hf   = (lane >= 16) ? 1 : 0;
    const int ln15 = lane & 15;
    const int qr0  = blockIdx.y * 64 + w * 16;

    const unsigned* Q32 = (const unsigned*)Q;

    Frag qf[2];
    #pragma unroll
    for (int c = 0; c < 2; ++c)
        #pragma unroll
        for (int i = 0; i < 8; ++i) {
            int d = c * 32 + a_koff(i, hf);
            qf[c].u[i] = Q32[(((size_t)(b * Ss + qr0 + ln15) * Hd + h * HDd + d) >> 1)];
        }

    float rm = -3.0e38f, rl = 0.f;
    v8f oacc[4];
    #pragma unroll
    for (int j = 0; j < 4; ++j) oacc[j] = vzero8();

    #pragma unroll 1
    for (int kt = 0; kt < Ss / 64; ++kt) {
        const int kv0 = kt * 64;
        // async-stage K [kv][d]: 64 rows x 128B = 512 16B chunks, 4 per thread
        #pragma unroll
        for (int p = 0; p < 4; ++p) {
            int cch = tid + p * 128;
            int row = cch >> 3, cp = cch & 7;
            async_cp16(Kb + (size_t)(b * Ss + kv0 + row) * Hd + h * HDd + cp * 8,
                       &k_s[row * 64 + cp * 8]);
        }
        // stage V transposed [d][kv] (VALU path; async cannot transpose)
        for (int e = tid; e < 4096; e += 128) {
            int kv = e >> 6, d = e & 63;
            v_s[d * 64 + kv] = Vb[(size_t)(b * Ss + kv0 + kv) * Hd + h * HDd + d];
        }
        if (kt + 1 < Ss / 64)
            __builtin_prefetch(Kb + (size_t)(b * Ss + kv0 + 64 + (tid >> 1)) * Hd + h * HDd, 0, 3);
        WAIT_ASYNC(0);
        __syncthreads();

        const unsigned* k32 = (const unsigned*)k_s;
        const unsigned* v32 = (const unsigned*)v_s;

        // S = Q K^T * 1/sqrt(64)
        #pragma unroll
        for (int j = 0; j < 4; ++j) {
            v8f sacc = vzero8();
            #pragma unroll
            for (int c = 0; c < 2; ++c) {
                Frag bf;
                int key = j * 16 + ln15;
                #pragma unroll
                for (int i = 0; i < 8; ++i) {
                    int dk = c * 32 + b_koff(i, hf);
                    bf.u[i] = k32[key * 32 + (dk >> 1)];
                }
                sacc = wmma_bf16(qf[c], bf, sacc);
            }
            #pragma unroll
            for (int i = 0; i < 8; ++i)
                s_s[w][(i + 8 * hf) * 64 + j * 16 + ln15] = sacc[i] * 0.125f;
        }
        __syncthreads();

        // online softmax: one row per lane (lanes 0..15)
        if (lane < 16) {
            const float* srow = &s_s[w][lane * 64];
            float tmax = srow[0];
            for (int t = 1; t < 64; ++t) tmax = fmaxf(tmax, srow[t]);
            float mnew  = fmaxf(rm, tmax);
            float scale = __expf(rm - mnew);
            float psum  = 0.f;
            unsigned short* prow = ((unsigned short*)p_s[w]) + lane * 64;
            for (int t = 0; t < 64; ++t) {
                float e = __expf(srow[t] - mnew);
                psum += e;
                prow[t] = f2bf(e);
            }
            rl = rl * scale + psum;
            rm = mnew;
            rscale[w][lane] = scale;
        }
        __syncthreads();

        #pragma unroll
        for (int j = 0; j < 4; ++j)
            #pragma unroll
            for (int i = 0; i < 8; ++i)
                oacc[j][i] *= rscale[w][i + 8 * hf];

        Frag pf[2];
        #pragma unroll
        for (int c = 0; c < 2; ++c)
            #pragma unroll
            for (int i = 0; i < 8; ++i) {
                int kk = c * 32 + a_koff(i, hf);
                pf[c].u[i] = p_s[w][(ln15 * 64 + kk) >> 1];
            }
        #pragma unroll
        for (int j = 0; j < 4; ++j) {
            #pragma unroll
            for (int c = 0; c < 2; ++c) {
                Frag vbf;
                int d = j * 16 + ln15;
                #pragma unroll
                for (int i = 0; i < 8; ++i) {
                    int kk = c * 32 + b_koff(i, hf);
                    vbf.u[i] = v32[(d * 64 + kk) >> 1];
                }
                oacc[j] = wmma_bf16(pf[c], vbf, oacc[j]);
            }
        }
        __syncthreads();
    }

    if (lane < 16) rinvl[w][lane] = 1.0f / rl;
    __syncthreads();
    #pragma unroll
    for (int j = 0; j < 4; ++j)
        #pragma unroll
        for (int i = 0; i < 8; ++i) {
            int m = i + 8 * hf;
            int row = qr0 + m;
            int d = j * 16 + ln15;
            out[(size_t)(b * Ss + row) * Hd + h * HDd + d] = oacc[j][i] * rinvl[w][m];
        }
}

// ---------------- Adapter: out += gelu(out @ Wd + bd) @ Wu + bu ----------------
// Wdt: [ADP][Hd] bf16 (= Wd^T), Wut: [Hd][ADP] bf16 (= Wu^T)
__global__ __launch_bounds__(128)
void adapter_kernel(float* __restrict__ out,
                    const unsigned short* __restrict__ Wdt, const float* __restrict__ bd,
                    const unsigned short* __restrict__ Wut, const float* __restrict__ bu) {
    __shared__ __align__(16) unsigned short a_s[64 * 32];   // [row][k]
    __shared__ __align__(16) unsigned short wd_s[64 * 32];  // [n][k]
    __shared__ __align__(16) unsigned short h_s[64 * 64];   // [row][k]
    __shared__ __align__(16) unsigned short wu_s[16 * 64];  // [n][k]

    const int tid  = threadIdx.x;
    const int lane = tid & 31;
    const int w    = tid >> 5;
    const int hf   = (lane >= 16) ? 1 : 0;
    const int ln15 = lane & 15;
    const int r0   = blockIdx.x * 64;

    v8f hacc[4];
    #pragma unroll
    for (int j = 0; j < 4; ++j) hacc[j] = vzero8();

    // down projection: [64,1024] @ [1024,64]
    #pragma unroll 1
    for (int k0 = 0; k0 < Hd; k0 += 32) {
        // A from fp32 attention output (convert)
        for (int e = tid; e < 1024; e += 128) {
            int row = e >> 4, kp = e & 15;
            float a0 = out[(size_t)(r0 + row) * Hd + k0 + 2 * kp];
            float a1 = out[(size_t)(r0 + row) * Hd + k0 + 2 * kp + 1];
            ((unsigned*)a_s)[row * 16 + kp] =
                (unsigned)f2bf(a0) | ((unsigned)f2bf(a1) << 16);
        }
        // Wd^T tile: 64 rows x 64B = 256 chunks, 2 per thread (async)
        #pragma unroll
        for (int p = 0; p < 2; ++p) {
            int cch = tid + p * 128;
            int n = cch >> 2, cp = cch & 3;
            async_cp16(Wdt + (size_t)n * Hd + k0 + cp * 8, &wd_s[n * 32 + cp * 8]);
        }
        WAIT_ASYNC(0);
        __syncthreads();
        const unsigned* la = (const unsigned*)a_s;
        const unsigned* lw = (const unsigned*)wd_s;
        Frag af;
        int row = w * 16 + ln15;
        #pragma unroll
        for (int i = 0; i < 8; ++i)
            af.u[i] = la[row * 16 + (a_koff(i, hf) >> 1)];
        #pragma unroll
        for (int j = 0; j < 4; ++j) {
            Frag bf;
            int n = j * 16 + ln15;
            #pragma unroll
            for (int i = 0; i < 8; ++i)
                bf.u[i] = lw[n * 16 + (b_koff(i, hf) >> 1)];
            hacc[j] = wmma_bf16(af, bf, hacc[j]);
        }
        __syncthreads();
    }

    // gelu (tanh approx, jax default) -> LDS bf16
    #pragma unroll
    for (int j = 0; j < 4; ++j)
        #pragma unroll
        for (int i = 0; i < 8; ++i) {
            int n = j * 16 + ln15;
            float x = hacc[j][i] + bd[n];
            float g = 0.5f * x * (1.0f + tanhf(0.7978845608f * (x + 0.044715f * x * x * x)));
            h_s[(w * 16 + i + 8 * hf) * 64 + n] = f2bf(g);
        }
    __syncthreads();

    const unsigned* lh = (const unsigned*)h_s;
    Frag hfr[2];
    {
        int row = w * 16 + ln15;
        #pragma unroll
        for (int c = 0; c < 2; ++c)
            #pragma unroll
            for (int i = 0; i < 8; ++i) {
                int k = c * 32 + a_koff(i, hf);
                hfr[c].u[i] = lh[(row * 64 + k) >> 1];
            }
    }

    #pragma unroll 1
    for (int jn = 0; jn < Hd / 16; ++jn) {
        const int n0 = jn * 16;
        // Wu^T tile: 16 rows x 128B = 128 chunks, 1 per thread (async)
        {
            int n = tid >> 3, cp = tid & 7;
            async_cp16(Wut + (size_t)(n0 + n) * ADP + cp * 8, &wu_s[n * 64 + cp * 8]);
        }
        if (jn + 1 < Hd / 16)
            __builtin_prefetch(Wut + (size_t)(n0 + 16 + (tid >> 3)) * ADP, 0, 3);
        WAIT_ASYNC(0);
        __syncthreads();
        const unsigned* lu = (const unsigned*)wu_s;
        v8f uacc = vzero8();
        #pragma unroll
        for (int c = 0; c < 2; ++c) {
            Frag bf;
            #pragma unroll
            for (int i = 0; i < 8; ++i) {
                int k = c * 32 + b_koff(i, hf);
                bf.u[i] = lu[ln15 * 32 + (k >> 1)];
            }
            uacc = wmma_bf16(hfr[c], bf, uacc);
        }
        #pragma unroll
        for (int i = 0; i < 8; ++i) {
            int row = r0 + w * 16 + i + 8 * hf;
            int col = n0 + ln15;
            out[(size_t)row * Hd + col] += uacc[i] + bu[col];
        }
        __syncthreads();
    }
}

extern "C" void kernel_launch(void* const* d_in, const int* in_sizes, int n_in,
                              void* d_out, int out_size, void* d_ws, size_t ws_size,
                              hipStream_t stream) {
    const float* hs  = (const float*)d_in[0];
    const float* ctx = (const float*)d_in[1];
    const float* Wq  = (const float*)d_in[2];
    const float* bq  = (const float*)d_in[3];
    const float* Wk  = (const float*)d_in[4];
    const float* bk  = (const float*)d_in[5];
    const float* Wv  = (const float*)d_in[6];
    const float* bv  = (const float*)d_in[7];
    const float* qg  = (const float*)d_in[8];
    const float* qb  = (const float*)d_in[9];
    const float* cg  = (const float*)d_in[10];
    const float* cb  = (const float*)d_in[11];
    const float* Wd  = (const float*)d_in[12];
    const float* bd  = (const float*)d_in[13];
    const float* Wu  = (const float*)d_in[14];
    const float* bu  = (const float*)d_in[15];
    float* out = (float*)d_out;

    const size_t MAT = (size_t)Mm * Hd;      // 8192*1024 elements
    unsigned short* hsn  = (unsigned short*)d_ws;
    unsigned short* ctxn = hsn  + MAT;
    unsigned short* Qb   = ctxn + MAT;
    unsigned short* Kb   = Qb   + MAT;
    unsigned short* Vb   = Kb   + MAT;
    unsigned short* Wqt  = Vb   + MAT;                    // [1024][1024]
    unsigned short* Wkt  = Wqt  + (size_t)Hd * Hd;
    unsigned short* Wvt  = Wkt  + (size_t)Hd * Hd;
    unsigned short* Wdt  = Wvt  + (size_t)Hd * Hd;        // [64][1024]
    unsigned short* Wut  = Wdt  + (size_t)ADP * Hd;       // [1024][64]

    ln_bf16_kernel<<<Mm, 256, 0, stream>>>(hs,  qg, qb, hsn);
    ln_bf16_kernel<<<Mm, 256, 0, stream>>>(ctx, cg, cb, ctxn);

    transpose_conv_kernel<<<dim3(16, 16), 256, 0, stream>>>(Wq, Wqt, Hd, Hd);
    transpose_conv_kernel<<<dim3(16, 16), 256, 0, stream>>>(Wk, Wkt, Hd, Hd);
    transpose_conv_kernel<<<dim3(16, 16), 256, 0, stream>>>(Wv, Wvt, Hd, Hd);
    transpose_conv_kernel<<<dim3(1, 16),  256, 0, stream>>>(Wd, Wdt, Hd, ADP);
    transpose_conv_kernel<<<dim3(16, 1),  256, 0, stream>>>(Wu, Wut, ADP, Hd);

    dim3 gg(Mm / 128, Hd / 128);
    gemm_bias_bf16_kernel<<<gg, 256, 0, stream>>>(hsn,  Wqt, bq, Qb);
    gemm_bias_bf16_kernel<<<gg, 256, 0, stream>>>(ctxn, Wkt, bk, Kb);
    gemm_bias_bf16_kernel<<<gg, 256, 0, stream>>>(ctxn, Wvt, bv, Vb);

    attention_kernel<<<dim3(Bb * NHh, Ss / 64), 128, 0, stream>>>(Qb, Kb, Vb, out);

    adapter_kernel<<<Mm / 64, 128, 0, stream>>>(out, Wdt, bd, Wut, bu);
}
